// ActionExpertSelfBlock_78056735637553
// MI455X (gfx1250) — compile-verified, matrix operand order checked by reading, not run on hardware
//
#include <hip/hip_runtime.h>

// ---------------------------------------------------------------------------
// CDNA5 (gfx1250, wave32) transformer block: v_wmma_f32_16x16x32_bf16 +
// global_load_async_to_lds_b128 double-buffered GEMM staging.
// ---------------------------------------------------------------------------

#define B_   2
#define L_   2048
#define D_   1024
#define QH_  8
#define KVH_ 2
#define HD_  128
#define FFN_ 4096
#define TOK_ (B_ * L_)   // 4096

typedef __attribute__((ext_vector_type(16))) __bf16 v16bf;
typedef __attribute__((ext_vector_type(8)))  __bf16 v8bf;
typedef __attribute__((ext_vector_type(8)))  float  v8f;

__device__ __forceinline__ v16bf cat8(v8bf a, v8bf b) {
    return __builtin_shufflevector(a, b, 0,1,2,3,4,5,6,7,8,9,10,11,12,13,14,15);
}

// LDS byte offset of a __shared__ object (ptrtoint of an addrspace(3) pointer)
__device__ __forceinline__ unsigned lds_off(const void* p) {
    return (unsigned)(unsigned long long)
        (__attribute__((address_space(3))) const char*)p;
}

// CDNA5 async global->LDS copy, 16 bytes per lane (GVS addressing mode).
// Tracked by ASYNCcnt; wait with s_wait_asynccnt.
__device__ __forceinline__ void async_ld_b128(unsigned ldsaddr, const void* base,
                                              int byteoff) {
    asm volatile("global_load_async_to_lds_b128 %0, %1, %2"
                 :: "v"(ldsaddr), "v"(byteoff), "s"(base)
                 : "memory");
}
__device__ __forceinline__ void wait_async0() {
    asm volatile("s_wait_asynccnt 0" ::: "memory");
}

// ---------------------------------------------------------------------------
// Weight convert + transpose: f32 in[K][N] (row-major) -> bf16 out[N][K]
// ---------------------------------------------------------------------------
__global__ void wcast_t(const float* __restrict__ in, __bf16* __restrict__ out,
                        int K, int N) {
    int idx = blockIdx.x * 256 + threadIdx.x;
    if (idx >= K * N) return;
    int k = idx % K, n = idx / K;
    out[(size_t)n * K + k] = (__bf16)in[(size_t)k * N + n];
}

// ---------------------------------------------------------------------------
// RMSNorm (f32 in -> bf16 out), one row (1024 cols) per 256-thread block
// ---------------------------------------------------------------------------
__global__ __launch_bounds__(256)
void rmsnorm_kernel(const float* __restrict__ x, const float* __restrict__ w,
                    __bf16* __restrict__ out) {
    __shared__ float red[256];
    const int row = blockIdx.x;
    const int t = threadIdx.x;
    const float4 xv = ((const float4*)(x + (size_t)row * D_))[t];
    red[t] = xv.x*xv.x + xv.y*xv.y + xv.z*xv.z + xv.w*xv.w;
    __syncthreads();
    for (int o = 128; o > 0; o >>= 1) {
        if (t < o) red[t] += red[t + o];
        __syncthreads();
    }
    const float r = rsqrtf(red[0] * (1.0f / D_) + 1.1920929e-07f);
    const float4 wv = ((const float4*)w)[t];
    __bf16* op = out + (size_t)row * D_ + t * 4;
    op[0] = (__bf16)(xv.x * r * wv.x);
    op[1] = (__bf16)(xv.y * r * wv.y);
    op[2] = (__bf16)(xv.z * r * wv.z);
    op[3] = (__bf16)(xv.w * r * wv.w);
}

// ---------------------------------------------------------------------------
// RoPE + cast to bf16. Layout [tok][H][128]; thread handles one (d, d+64) pair.
// ---------------------------------------------------------------------------
__global__ void rope_cast(const float* __restrict__ in, __bf16* __restrict__ out,
                          int H, float scale) {
    int idx = blockIdx.x * 256 + threadIdx.x;
    if (idx >= B_ * L_ * H * 64) return;
    const int d   = idx & 63;
    const int h   = (idx >> 6) % H;
    const int tok = idx / (64 * H);
    const int pos = tok % L_;
    const float rad = (float)pos * __expf(-0.14391156827f * (float)d);
    float s, c;
    __sincosf(rad, &s, &c);
    const size_t base = ((size_t)tok * H + h) * HD_;
    const float x1 = in[base + d], x2 = in[base + 64 + d];
    out[base + d]      = (__bf16)((x1 * c - x2 * s) * scale);
    out[base + 64 + d] = (__bf16)((x2 * c + x1 * s) * scale);
}

// ---------------------------------------------------------------------------
// V transpose: [b][l][kvh][hd] -> [b][kvh][hd][L]
// ---------------------------------------------------------------------------
__global__ void vtrans(const __bf16* __restrict__ in, __bf16* __restrict__ out) {
    int idx = blockIdx.x * 256 + threadIdx.x;   // total = B*KVH*HD*L = 1M
    const int l = idx % L_;
    const int d = (idx / L_) % HD_;
    const int h = (idx / (L_ * HD_)) % KVH_;
    const int b = idx / (L_ * HD_ * KVH_);
    out[idx] = in[(((size_t)b * L_ + l) * KVH_ + h) * HD_ + d];
}

// ---------------------------------------------------------------------------
// Tiled WMMA GEMM: C[M,N] = A[M,K] @ B[K,N], A row-major bf16, Bt[N][K] bf16.
// Block = 128x128, BK=32, 256 threads = 8 waves; each wave = 32x64 = 8 WMMA
// tiles. LDS double-buffered; tiles staged with global_load_async_to_lds_b128
// (ASYNCcnt) so the next K-step's fill overlaps this K-step's WMMAs.
// EPI: 0 = f32 store, 1 = bf16 store, 2 = f32 + residual(aux),
//      3 = bf16 store of silu(aux)*C  (fused SwiGLU)
// ---------------------------------------------------------------------------
template <int EPI>
__global__ __launch_bounds__(256)
void gemm_bf16(const __bf16* __restrict__ A, const __bf16* __restrict__ Bt,
               void* __restrict__ outp, const float* __restrict__ aux,
               int M, int N, int K) {
    __shared__ __align__(16) __bf16 As[2][128][32];
    __shared__ __align__(16) __bf16 Bs[2][128][32];
    const int tid  = threadIdx.x;
    const int lane = tid & 31;
    const int w    = tid >> 5;
    const int wm   = w & 3;            // 0..3 -> 32-row M sub-tile
    const int wn   = w >> 2;           // 0..1 -> 64-col N half
    const int m0   = blockIdx.y * 128;
    const int n0   = blockIdx.x * 128;
    const int lr   = lane & 15;
    const int half = (lane < 16) ? 0 : 8;   // K-half per 16-bit operand layout

    // per-thread fill assignment: 128 rows x 32 cols, 16 halves (32B) each
    const int frow = tid >> 1;          // 0..127
    const int fcol = (tid & 1) * 16;    // 0 or 16
    const unsigned ldsA[2] = { lds_off(&As[0][frow][fcol]),
                               lds_off(&As[1][frow][fcol]) };
    const unsigned ldsB[2] = { lds_off(&Bs[0][frow][fcol]),
                               lds_off(&Bs[1][frow][fcol]) };
    const int gA = ((m0 + frow) * K + fcol) * 2;   // byte offsets
    const int gB = ((n0 + frow) * K + fcol) * 2;

    v8f acc[2][4] = {};

    // prologue: fill buffer 0 for kt = 0
    async_ld_b128(ldsA[0],      A,  gA);
    async_ld_b128(ldsA[0] + 16, A,  gA + 16);
    async_ld_b128(ldsB[0],      Bt, gB);
    async_ld_b128(ldsB[0] + 16, Bt, gB + 16);
    wait_async0();
    __syncthreads();

    int cur = 0;
    for (int kt = 0; kt < K; kt += 32) {
        const int nxt = cur ^ 1;
        if (kt + 32 < K) {
            const int go = (kt + 32) * 2;
            async_ld_b128(ldsA[nxt],      A,  gA + go);
            async_ld_b128(ldsA[nxt] + 16, A,  gA + go + 16);
            async_ld_b128(ldsB[nxt],      Bt, gB + go);
            async_ld_b128(ldsB[nxt] + 16, Bt, gB + go + 16);
        }
        // A fragments: 2 M-tiles of 16
        v16bf af[2];
#pragma unroll
        for (int i = 0; i < 2; ++i) {
            const int r = wm * 32 + i * 16 + lr;
            af[i] = cat8(*(const v8bf*)&As[cur][r][half],
                         *(const v8bf*)&As[cur][r][half + 16]);
        }
#pragma unroll
        for (int t = 0; t < 4; ++t) {
            const int nrow = wn * 64 + t * 16 + lr;
            const v16bf bfv = cat8(*(const v8bf*)&Bs[cur][nrow][half],
                                   *(const v8bf*)&Bs[cur][nrow][half + 16]);
#pragma unroll
            for (int i = 0; i < 2; ++i)
                acc[i][t] = __builtin_amdgcn_wmma_f32_16x16x32_bf16(
                    false, af[i], false, bfv, (short)0, acc[i][t], false, false);
        }
        wait_async0();
        __syncthreads();
        cur = nxt;
    }

    // C layout: lanes 0-15 -> rows v, cols lane; lanes 16-31 -> rows v+8
    const int rb = m0 + wm * 32 + ((lane >> 4) ? 8 : 0);
#pragma unroll
    for (int i = 0; i < 2; ++i) {
#pragma unroll
        for (int t = 0; t < 4; ++t) {
            const int col = n0 + wn * 64 + t * 16 + lr;
#pragma unroll
            for (int v = 0; v < 8; ++v) {
                const size_t idx = (size_t)(rb + i * 16 + v) * N + col;
                const float c = acc[i][t][v];
                if (EPI == 0) {
                    ((float*)outp)[idx] = c;
                } else if (EPI == 1) {
                    ((__bf16*)outp)[idx] = (__bf16)c;
                } else if (EPI == 2) {
                    ((float*)outp)[idx] = c + aux[idx];
                } else {
                    const float g = aux[idx];
                    const float sg = g / (1.0f + __expf(-g));
                    ((__bf16*)outp)[idx] = (__bf16)(sg * c);
                }
            }
        }
    }
}

// ---------------------------------------------------------------------------
// Flash-style causal attention, one wave per 16-query tile (unchanged:
// already pipelines global_load_b128 against v_wmma well; K/V live in L2).
// ---------------------------------------------------------------------------
__global__ __launch_bounds__(32)
void attn_kernel(const __bf16* __restrict__ q, const __bf16* __restrict__ k,
                 const __bf16* __restrict__ vt, __bf16* __restrict__ ctx) {
    __shared__ __align__(16) __bf16 ps[16][32];
    const int lane = threadIdx.x;
    const int lr   = lane & 15;
    const int hi   = lane >> 4;
    const int half = hi ? 8 : 0;
    const int qs   = blockIdx.x * 16;
    const int h    = blockIdx.y;
    const int b    = blockIdx.z;
    const int kvh  = h / (QH_ / KVH_);

    const __bf16* qp = q + (((size_t)(b * L_ + qs + lr) * QH_) + h) * HD_;
    v16bf qf[4];
#pragma unroll
    for (int j = 0; j < 4; ++j)
        qf[j] = cat8(*(const v8bf*)&qp[32 * j + half],
                     *(const v8bf*)&qp[32 * j + half + 16]);

    v8f O[8] = {};
    float mrow[8], lrow[8];
#pragma unroll
    for (int v = 0; v < 8; ++v) { mrow[v] = -1e30f; lrow[v] = 0.f; }

    const int kend = qs + 16;
    for (int k0 = 0; k0 < kend; k0 += 32) {
        v8f s0 = {}, s1 = {};
        const __bf16* kp0 = k + ((size_t)(b * L_ + k0 + lr) * KVH_ + kvh) * HD_;
        const __bf16* kp1 = kp0 + (size_t)16 * KVH_ * HD_;
#pragma unroll
        for (int j = 0; j < 4; ++j) {
            const v16bf kf0 = cat8(*(const v8bf*)&kp0[32 * j + half],
                                   *(const v8bf*)&kp0[32 * j + half + 16]);
            const v16bf kf1 = cat8(*(const v8bf*)&kp1[32 * j + half],
                                   *(const v8bf*)&kp1[32 * j + half + 16]);
            s0 = __builtin_amdgcn_wmma_f32_16x16x32_bf16(
                false, qf[j], false, kf0, (short)0, s0, false, false);
            s1 = __builtin_amdgcn_wmma_f32_16x16x32_bf16(
                false, qf[j], false, kf1, (short)0, s1, false, false);
        }
#pragma unroll
        for (int v = 0; v < 8; ++v) {
            const int qrow = qs + v + 8 * hi;
            if (k0 + lr > qrow)      s0[v] = -1e30f;
            if (k0 + 16 + lr > qrow) s1[v] = -1e30f;
        }
        float al[8];
#pragma unroll
        for (int v = 0; v < 8; ++v) {
            float mx = fmaxf(s0[v], s1[v]);
            mx = fmaxf(mx, __shfl_xor(mx, 1, 32));
            mx = fmaxf(mx, __shfl_xor(mx, 2, 32));
            mx = fmaxf(mx, __shfl_xor(mx, 4, 32));
            mx = fmaxf(mx, __shfl_xor(mx, 8, 32));
            const float mn = fmaxf(mrow[v], mx);
            al[v] = __expf(mrow[v] - mn);
            mrow[v] = mn;
        }
#pragma unroll
        for (int v = 0; v < 8; ++v) {
            const float p0 = __expf(s0[v] - mrow[v]);
            const float p1 = __expf(s1[v] - mrow[v]);
            const int row = v + 8 * hi;
            ps[row][lr]      = (__bf16)p0;
            ps[row][16 + lr] = (__bf16)p1;
            float sm = p0 + p1;
            sm += __shfl_xor(sm, 1, 32);
            sm += __shfl_xor(sm, 2, 32);
            sm += __shfl_xor(sm, 4, 32);
            sm += __shfl_xor(sm, 8, 32);
            lrow[v] = lrow[v] * al[v] + sm;
        }
        __syncthreads();
        const v16bf pf = cat8(*(const v8bf*)&ps[lr][half],
                              *(const v8bf*)&ps[lr][half + 16]);
        v8f av;
#pragma unroll
        for (int v = 0; v < 8; ++v) av[v] = al[v];
        const __bf16* vbase = vt + ((size_t)(b * KVH_ + kvh) * HD_) * L_ + k0;
#pragma unroll
        for (int f = 0; f < 8; ++f) {
            const __bf16* vp = vbase + (size_t)(16 * f + lr) * L_;
            const v16bf vf = cat8(*(const v8bf*)&vp[half],
                                  *(const v8bf*)&vp[half + 16]);
            O[f] = O[f] * av;
            O[f] = __builtin_amdgcn_wmma_f32_16x16x32_bf16(
                false, pf, false, vf, (short)0, O[f], false, false);
        }
        __syncthreads();
    }
    float rinv[8];
#pragma unroll
    for (int v = 0; v < 8; ++v) rinv[v] = 1.0f / lrow[v];
#pragma unroll
    for (int f = 0; f < 8; ++f) {
#pragma unroll
        for (int v = 0; v < 8; ++v) {
            const int row = qs + v + 8 * hi;
            const int col = 16 * f + lr;
            ctx[(((size_t)(b * L_ + row) * QH_) + h) * HD_ + col] =
                (__bf16)(O[f][v] * rinv[v]);
        }
    }
}

// ---------------------------------------------------------------------------
// Host launcher
// ---------------------------------------------------------------------------
extern "C" void kernel_launch(void* const* d_in, const int* in_sizes, int n_in,
                              void* d_out, int out_size, void* d_ws, size_t ws_size,
                              hipStream_t stream) {
    (void)in_sizes; (void)n_in; (void)out_size; (void)ws_size;
    const float* x    = (const float*)d_in[0];
    const float* ln1w = (const float*)d_in[1];
    const float* wq   = (const float*)d_in[2];
    const float* wk   = (const float*)d_in[3];
    const float* wv   = (const float*)d_in[4];
    const float* wo   = (const float*)d_in[5];
    const float* ln2w = (const float*)d_in[6];
    const float* wg   = (const float*)d_in[7];
    const float* wu   = (const float*)d_in[8];
    const float* wd   = (const float*)d_in[9];
    float* out = (float*)d_out;

    char* p = (char*)d_ws;
    auto alloc = [&](size_t bytes) {
        char* r = p;
        p += (bytes + 255) & ~(size_t)255;
        return r;
    };
    __bf16* wq_t = (__bf16*)alloc((size_t)D_ * (QH_ * HD_) * 2);
    __bf16* wk_t = (__bf16*)alloc((size_t)D_ * (KVH_ * HD_) * 2);
    __bf16* wv_t = (__bf16*)alloc((size_t)D_ * (KVH_ * HD_) * 2);
    __bf16* wo_t = (__bf16*)alloc((size_t)(QH_ * HD_) * D_ * 2);
    __bf16* wg_t = (__bf16*)alloc((size_t)D_ * FFN_ * 2);
    __bf16* wu_t = (__bf16*)alloc((size_t)D_ * FFN_ * 2);
    __bf16* wd_t = (__bf16*)alloc((size_t)FFN_ * D_ * 2);
    __bf16* h1   = (__bf16*)alloc((size_t)TOK_ * D_ * 2);
    float*  qtmp = (float*) alloc((size_t)TOK_ * QH_ * HD_ * 4);
    float*  ktmp = (float*) alloc((size_t)TOK_ * KVH_ * HD_ * 4);
    __bf16* vbuf = (__bf16*)alloc((size_t)TOK_ * KVH_ * HD_ * 2);
    __bf16* qro  = (__bf16*)alloc((size_t)TOK_ * QH_ * HD_ * 2);
    __bf16* kro  = (__bf16*)alloc((size_t)TOK_ * KVH_ * HD_ * 2);
    __bf16* vt   = (__bf16*)alloc((size_t)B_ * KVH_ * HD_ * L_ * 2);
    __bf16* ctx  = (__bf16*)alloc((size_t)TOK_ * QH_ * HD_ * 2);
    float*  x1   = (float*) alloc((size_t)TOK_ * D_ * 4);
    __bf16* h2   = (__bf16*)alloc((size_t)TOK_ * D_ * 2);
    float*  gbuf = (float*) alloc((size_t)TOK_ * FFN_ * 4);
    __bf16* gu   = (__bf16*)alloc((size_t)TOK_ * FFN_ * 2);

    auto cdiv = [](int a, int b) { return (a + b - 1) / b; };

    // 1) weights -> bf16 transposed [N][K]
    wcast_t<<<cdiv(D_ * QH_ * HD_, 256), 256, 0, stream>>>(wq, wq_t, D_, QH_ * HD_);
    wcast_t<<<cdiv(D_ * KVH_ * HD_, 256), 256, 0, stream>>>(wk, wk_t, D_, KVH_ * HD_);
    wcast_t<<<cdiv(D_ * KVH_ * HD_, 256), 256, 0, stream>>>(wv, wv_t, D_, KVH_ * HD_);
    wcast_t<<<cdiv(QH_ * HD_ * D_, 256), 256, 0, stream>>>(wo, wo_t, QH_ * HD_, D_);
    wcast_t<<<cdiv(D_ * FFN_, 256), 256, 0, stream>>>(wg, wg_t, D_, FFN_);
    wcast_t<<<cdiv(D_ * FFN_, 256), 256, 0, stream>>>(wu, wu_t, D_, FFN_);
    wcast_t<<<cdiv(FFN_ * D_, 256), 256, 0, stream>>>(wd, wd_t, FFN_, D_);

    // 2) RMSNorm1
    rmsnorm_kernel<<<TOK_, 256, 0, stream>>>(x, ln1w, h1);

    // 3) QKV projections
    gemm_bf16<0><<<dim3((QH_ * HD_) / 128, TOK_ / 128), 256, 0, stream>>>(
        h1, wq_t, qtmp, nullptr, TOK_, QH_ * HD_, D_);
    gemm_bf16<0><<<dim3((KVH_ * HD_) / 128, TOK_ / 128), 256, 0, stream>>>(
        h1, wk_t, ktmp, nullptr, TOK_, KVH_ * HD_, D_);
    gemm_bf16<1><<<dim3((KVH_ * HD_) / 128, TOK_ / 128), 256, 0, stream>>>(
        h1, wv_t, vbuf, nullptr, TOK_, KVH_ * HD_, D_);

    // 4) RoPE (Q pre-scaled by 1/sqrt(HD)); V transpose
    rope_cast<<<cdiv(B_ * L_ * QH_ * 64, 256), 256, 0, stream>>>(
        qtmp, qro, QH_, 0.08838834764831845f);
    rope_cast<<<cdiv(B_ * L_ * KVH_ * 64, 256), 256, 0, stream>>>(
        ktmp, kro, KVH_, 1.0f);
    vtrans<<<cdiv(B_ * KVH_ * HD_ * L_, 256), 256, 0, stream>>>(vbuf, vt);

    // 5) causal flash attention
    attn_kernel<<<dim3(L_ / 16, QH_, B_), 32, 0, stream>>>(qro, kro, vt, ctx);

    // 6) output projection + residual
    gemm_bf16<2><<<dim3(D_ / 128, TOK_ / 128), 256, 0, stream>>>(
        ctx, wo_t, x1, x, TOK_, D_, QH_ * HD_);

    // 7) RMSNorm2
    rmsnorm_kernel<<<TOK_, 256, 0, stream>>>(x1, ln2w, h2);

    // 8) SwiGLU MLP
    gemm_bf16<0><<<dim3(FFN_ / 128, TOK_ / 128), 256, 0, stream>>>(
        h2, wg_t, gbuf, nullptr, TOK_, FFN_, D_);
    gemm_bf16<3><<<dim3(FFN_ / 128, TOK_ / 128), 256, 0, stream>>>(
        h2, wu_t, gu, gbuf, TOK_, FFN_, D_);
    gemm_bf16<2><<<dim3(D_ / 128, TOK_ / 128), 256, 0, stream>>>(
        gu, wd_t, out, x1, TOK_, D_, FFN_);
}
